// CustomModel_34162169872800
// MI455X (gfx1250) — matrix operand order repaired and checked by reading, not compiled, not run on hardware
//
#include <hip/hip_runtime.h>

typedef _Float16 half_t;
typedef __attribute__((ext_vector_type(16))) _Float16 v16h;
typedef __attribute__((ext_vector_type(8)))  _Float16 v8h;
typedef __attribute__((ext_vector_type(8)))  float    v8f;

#define DEV static __device__ __forceinline__

constexpr int BB = 32, SSQ = 200, DD = 768, FFD = 3072, LLN = 12, HHN = 12, DHD = 64;
constexpr int SP = 224;                       // S padded to multiple of 32 (K-pad for ctx GEMM)
constexpr int NTOK = BB * SSQ;                // 6400
constexpr long ATTN_ELEMS = (long)BB * HHN * SSQ * SSQ;   // 15,360,000
constexpr long ATT_ROWS   = (long)BB * HHN * SSQ;         // 76,800

// ---------------------------------------------------------------------------
// Unguarded WMMA fragment loaders (CDNA5 16x16x32 f16 layout).
// A 16x32: lanes 0-15 row M, K = {kk..kk+7, kk+16..kk+23}; lanes 16-31 K+8.
// B as B^T rows: lane n holds K = {kk..kk+7, kk+16..kk+23} + hl*8 (same as A).
// Callers guarantee in-buffer reads via slack rows / zero K-padding.
// ---------------------------------------------------------------------------
DEV v16h load_a16(const half_t* A, long rowoff, int kk, int hl) {
  const v8h* p = (const v8h*)(A + rowoff + kk + hl * 8);
  v8h lo = p[0];
  v8h hi = p[2];  // +16 halves
  return __builtin_shufflevector(lo, hi, 0, 1, 2, 3, 4, 5, 6, 7,
                                 8, 9, 10, 11, 12, 13, 14, 15);
}

DEV v16h load_b16(const half_t* Bt, long rowoff, int kk, int hl) {
  const v8h* p = (const v8h*)(Bt + rowoff + kk + hl * 8);
  v8h lo = p[0];
  v8h hi = p[2];
  return __builtin_shufflevector(lo, hi, 0, 1, 2, 3, 4, 5, 6, 7,
                                 8, 9, 10, 11, 12, 13, 14, 15);
}

template <bool OUT16, bool EDGE>
DEV void store_acc(const v8f& acc, void* C, long base, int ldc, int m0, int n,
                   int M, int N, float bv, int act) {
#pragma unroll
  for (int r = 0; r < 8; r++) {
    int m = m0 + r;
    if (!EDGE || (m < M && n < N)) {
      float v = acc[r] + bv;
      if (act == 1)      v = 0.5f * v * (1.f + erff(v * 0.70710678118f)); // exact GELU
      else if (act == 2) v = fmaxf(v, 0.f);
      if (OUT16) ((half_t*)C)[base + (long)m * ldc + n] = (half_t)v;
      else       ((float*)C)[base + (long)m * ldc + n]  = v;
    }
  }
}

// ---------------------------------------------------------------------------
// WMMA GEMM: C = act(A * Bt^T + bias). One wave computes a 32x32 tile
// (4 accumulators; A/B fragments each reused twice). All loads unconditional;
// EDGE only masks stores. Batched via blockIdx.y (yb = y/Hs, yh = y%Hs).
// ---------------------------------------------------------------------------
template <bool OUT16, bool EDGE>
__global__ void gemm_wmma_kernel(const half_t* A, const half_t* Bt, const float* bias,
                                 void* Cout, int M, int N, int K,
                                 int lda, int ldb, int ldc, int Hs,
                                 long sAb, long sAh, long sBb, long sBh,
                                 long sCb, long sCh, int act) {
  int y = blockIdx.y;
  int yb = y / Hs, yh = y % Hs;
  const half_t* Ap = A + (long)yb * sAb + (long)yh * sAh;
  const half_t* Bp = Bt + (long)yb * sBb + (long)yh * sBh;
  long offC = (long)yb * sCb + (long)yh * sCh;

  int lane = threadIdx.x & 31;
  int wv   = threadIdx.x >> 5;
  int ntn  = (N + 31) >> 5;
  int ntm  = (M + 31) >> 5;
  int tile = blockIdx.x * (blockDim.x >> 5) + wv;
  if (tile >= ntm * ntn) return;   // wave-uniform: EXEC all-1 for WMMA
  int row0 = (tile / ntn) * 32;
  int col0 = (tile % ntn) * 32;
  int hl = lane >> 4, l15 = lane & 15;

  long ra0 = (long)(row0 + l15) * lda;
  long ra1 = (long)(row0 + 16 + l15) * lda;
  long rb0 = (long)(col0 + l15) * ldb;
  long rb1 = (long)(col0 + 16 + l15) * ldb;

  v8f acc00 = {}, acc01 = {}, acc10 = {}, acc11 = {};
  for (int kk = 0; kk < K; kk += 32) {
    v16h a0 = load_a16(Ap, ra0, kk, hl);
    v16h a1 = load_a16(Ap, ra1, kk, hl);
    v16h b0 = load_b16(Bp, rb0, kk, hl);
    v16h b1 = load_b16(Bp, rb1, kk, hl);
    acc00 = __builtin_amdgcn_wmma_f32_16x16x32_f16(false, a0, false, b0, (short)0, acc00, false, false);
    acc01 = __builtin_amdgcn_wmma_f32_16x16x32_f16(false, a0, false, b1, (short)0, acc01, false, false);
    acc10 = __builtin_amdgcn_wmma_f32_16x16x32_f16(false, a1, false, b0, (short)0, acc10, false, false);
    acc11 = __builtin_amdgcn_wmma_f32_16x16x32_f16(false, a1, false, b1, (short)0, acc11, false, false);
  }

  int n0 = col0 + l15, n1 = n0 + 16;
  float bv0 = 0.f, bv1 = 0.f;
  if (bias != nullptr) {
    bv0 = (!EDGE || n0 < N) ? bias[n0] : 0.f;
    bv1 = (!EDGE || n1 < N) ? bias[n1] : 0.f;
  }
  int m0 = row0 + hl * 8;
  store_acc<OUT16, EDGE>(acc00, Cout, offC, ldc, m0,      n0, M, N, bv0, act);
  store_acc<OUT16, EDGE>(acc01, Cout, offC, ldc, m0,      n1, M, N, bv1, act);
  store_acc<OUT16, EDGE>(acc10, Cout, offC, ldc, m0 + 16, n0, M, N, bv0, act);
  store_acc<OUT16, EDGE>(acc11, Cout, offC, ldc, m0 + 16, n1, M, N, bv1, act);
}

// ---------------------------------------------------------------------------
// Block reduction (wave32 shfl + LDS across 8 waves; blockDim == 256)
// ---------------------------------------------------------------------------
DEV void blk_reduce2(float& a, float& b) {
  __shared__ float r1[8], r2[8];
  int lane = threadIdx.x & 31, w = threadIdx.x >> 5;
#pragma unroll
  for (int o = 16; o > 0; o >>= 1) { a += __shfl_xor(a, o, 32); b += __shfl_xor(b, o, 32); }
  if (lane == 0) { r1[w] = a; r2[w] = b; }
  __syncthreads();
  if (threadIdx.x == 0) {
    float sa = 0.f, sb = 0.f;
    for (int i = 0; i < 8; i++) { sa += r1[i]; sb += r2[i]; }
    r1[0] = sa; r2[0] = sb;
  }
  __syncthreads();
  a = r1[0]; b = r2[0];
}

// ---------------------------------------------------------------------------
// Position ids (RoBERTa cumsum), embedding + LN, residual + LN
// ---------------------------------------------------------------------------
__global__ void pos_kernel(const int* ids, int* pos) {
  int i = blockIdx.x * blockDim.x + threadIdx.x;
  if (i >= BB * SSQ) return;
  int b = i / SSQ, s = i % SSQ;
  int cnt = 0;
  for (int j = 0; j <= s; j++) cnt += (ids[b * SSQ + j] != 1);
  pos[i] = (ids[i] != 1) ? (cnt + 1) : 1;
}

__global__ void embed_ln_kernel(const int* ids, const int* pos, const float* we,
                                const float* pe, const float* te, const float* g,
                                const float* be, float* x, half_t* x16) {
  long tok = blockIdx.x;
  int id = ids[tok], p = pos[tok];
  int t = threadIdx.x;
  float vals[3]; float s = 0.f, s2 = 0.f;
#pragma unroll
  for (int i = 0; i < 3; i++) {
    int d = t + i * 256;
    float v = we[(long)id * DD + d] + pe[(long)p * DD + d] + te[d];
    vals[i] = v; s += v; s2 += v * v;
  }
  blk_reduce2(s, s2);
  float mean = s / (float)DD;
  float rstd = rsqrtf(s2 / (float)DD - mean * mean + 1e-5f);
#pragma unroll
  for (int i = 0; i < 3; i++) {
    int d = t + i * 256;
    float o = (vals[i] - mean) * rstd * g[d] + be[d];
    x[tok * DD + d] = o;
    x16[tok * DD + d] = (half_t)o;
  }
}

__global__ void add_ln_kernel(const float* xin, const float* delta, const float* g,
                              const float* be, float* xout, half_t* x16) {
  long tok = blockIdx.x;
  int t = threadIdx.x;
  float vals[3]; float s = 0.f, s2 = 0.f;
#pragma unroll
  for (int i = 0; i < 3; i++) {
    int d = t + i * 256;
    float v = xin[tok * DD + d] + delta[tok * DD + d];
    vals[i] = v; s += v; s2 += v * v;
  }
  blk_reduce2(s, s2);
  float mean = s / (float)DD;
  float rstd = rsqrtf(s2 / (float)DD - mean * mean + 1e-5f);
#pragma unroll
  for (int i = 0; i < 3; i++) {
    int d = t + i * 256;
    float o = (vals[i] - mean) * rstd * g[d] + be[d];
    xout[tok * DD + d] = o;
    x16[tok * DD + d] = (half_t)o;
  }
}

// ---------------------------------------------------------------------------
// Masked scaled softmax over rows of att [rows, SP]; emits f32 (in place,
// cols < S) and f16 probs (ld = SP, padding cols stay zero-filled).
// ---------------------------------------------------------------------------
__global__ void softmax_mask_kernel(float* att, half_t* p16, const int* amask, float scale) {
  int row = blockIdx.x;                    // b*H*S + h*S + q
  int b = row / (HHN * SSQ);
  float* p = att + (long)row * SP;
  int t = threadIdx.x;
  float logit = 0.f, v = -INFINITY;
  if (t < SSQ) {
    logit = p[t] * scale + (amask[b * SSQ + t] ? 0.f : -1e9f);
    v = logit;
  }
  __shared__ float sm[8];
  float m = v;
#pragma unroll
  for (int o = 16; o > 0; o >>= 1) m = fmaxf(m, __shfl_xor(m, o, 32));
  if ((t & 31) == 0) sm[t >> 5] = m;
  __syncthreads();
  if (t < 8) {
    float mm = sm[t];
#pragma unroll
    for (int o = 4; o > 0; o >>= 1) mm = fmaxf(mm, __shfl_xor(mm, o, 32));
    if (t == 0) sm[0] = mm;
  }
  __syncthreads();
  m = sm[0];
  float e = (t < SSQ) ? __expf(logit - m) : 0.f;
  __shared__ float ssum[8];
  float s = e;
#pragma unroll
  for (int o = 16; o > 0; o >>= 1) s += __shfl_xor(s, o, 32);
  if ((t & 31) == 0) ssum[t >> 5] = s;
  __syncthreads();
  if (t < 8) {
    float t2 = ssum[t];
#pragma unroll
    for (int o = 4; o > 0; o >>= 1) t2 += __shfl_xor(t2, o, 32);
    if (t == 0) ssum[0] = t2;
  }
  __syncthreads();
  if (t < SSQ) {
    float pr = e / ssum[0];
    p[t] = pr;
    p16[(long)row * SP + t] = (half_t)pr;
  }
}

// ---------------------------------------------------------------------------
// Utility kernels
// ---------------------------------------------------------------------------
__global__ void zero_kernel(unsigned int* p, long nwords) {
  long i = (long)blockIdx.x * blockDim.x + threadIdx.x;
  if (i < nwords) p[i] = 0u;
}

__global__ void wt_f16_kernel(const float* W, half_t* WT, int K, int N, long total) {
  long i = (long)blockIdx.x * blockDim.x + threadIdx.x;
  if (i >= total) return;
  long per = (long)K * N;
  long l = i / per, r = i - l * per;
  int n = (int)(r / K), k = (int)(r - (long)n * K);
  WT[l * per + (long)n * K + k] = (half_t)W[l * per + (long)k * N + n];
}

// v [B,S,H*DH] (f16) -> vT [B,H,DH,SP] (padding cols untouched: pre-zeroed)
__global__ void vT_kernel(const half_t* v, half_t* vT) {
  long total = (long)BB * HHN * DHD * SSQ;
  long i = (long)blockIdx.x * blockDim.x + threadIdx.x;
  if (i >= total) return;
  int s = (int)(i % SSQ); long r = i / SSQ;
  int dh = (int)(r % DHD); r /= DHD;
  int h = (int)(r % HHN);  int b = (int)(r / HHN);
  vT[(((long)(b * HHN + h)) * DHD + dh) * SP + s] =
      v[((long)(b * SSQ + s)) * DD + h * DHD + dh];
}

__global__ void copy_att_kernel(const float* att, float* dst) {
  long i = (long)blockIdx.x * blockDim.x + threadIdx.x;
  if (i >= ATTN_ELEMS) return;
  long row = i / SSQ;
  int col = (int)(i % SSQ);
  dst[i] = att[row * SP + col];
}

__global__ void conv_relu_kernel(const float* x, const float* w, const float* bias,
                                 float* y, int NN, int Cin, int Win, int Cout) {
  int Wout = Win - 2;
  long total = (long)NN * Cout * Wout;
  long i = (long)blockIdx.x * blockDim.x + threadIdx.x;
  if (i >= total) return;
  int ow = (int)(i % Wout); long r = i / Wout;
  int co = (int)(r % Cout); int n = (int)(r / Cout);
  float s = bias[co];
  for (int ci = 0; ci < Cin; ci++) {
    const float* xp = x + ((long)n * Cin + ci) * Win + ow;
    const float* wp = w + ((long)co * Cin + ci) * 3;
    s += xp[0] * wp[0] + xp[1] * wp[1] + xp[2] * wp[2];
  }
  y[i] = fmaxf(s, 0.f);
}

__global__ void pool3_kernel(const float* x, float* y, long NC, int Win) {
  int Wout = Win / 3;
  long total = NC * Wout;
  long i = (long)blockIdx.x * blockDim.x + threadIdx.x;
  if (i >= total) return;
  int ow = (int)(i % Wout); long nc = i / Wout;
  const float* p = x + nc * Win + (long)ow * 3;
  y[i] = fmaxf(p[0], fmaxf(p[1], p[2]));
}

__global__ void head_kernel(const float* flat, const float* extra, const float* We,
                            const float* be, const float* Wc, const float* bc,
                            const float* Wout, const float* bout, float* out) {
  int b = blockIdx.x, t = threadIdx.x;
  __shared__ float ef[64];
  __shared__ float hid[128];
  if (t < 64) {
    float s = be[t];
    for (int i = 0; i < 10; i++) s += extra[b * 10 + i] * We[i * 64 + t];
    ef[t] = s;
  }
  __syncthreads();
  if (t < 128) {
    float s = bc[t];
    for (int i = 0; i < 432; i++) s += flat[b * 432 + i] * Wc[i * 128 + t];
    for (int i = 0; i < 64; i++)  s += ef[i] * Wc[(432 + i) * 128 + t];
    hid[t] = fmaxf(s, 0.f);
  }
  __syncthreads();
  if (t < 6) {
    float s = bout[t];
    for (int i = 0; i < 128; i++) s += hid[i] * Wout[i * 6 + t];
    out[b * 6 + t] = s;
    out[192 + b * 6 + t] = 1.f / (1.f + __expf(-s));
  }
}

// ---------------------------------------------------------------------------
// Host-side launch helper
// ---------------------------------------------------------------------------
static void launch_gemm(hipStream_t st, bool out16, bool edge,
                        const half_t* A, const half_t* Bt, const float* bias,
                        void* C, int M, int N, int K, int lda, int ldb, int ldc,
                        int act, int gy, int Hs, long sAb, long sAh,
                        long sBb, long sBh, long sCb, long sCh) {
  int tiles = ((M + 31) / 32) * ((N + 31) / 32);
  dim3 grid((tiles + 7) / 8, gy);
  if (out16) {
    if (edge) gemm_wmma_kernel<true, true><<<grid, 256, 0, st>>>(A, Bt, bias, C, M, N, K, lda, ldb, ldc, Hs, sAb, sAh, sBb, sBh, sCb, sCh, act);
    else      gemm_wmma_kernel<true, false><<<grid, 256, 0, st>>>(A, Bt, bias, C, M, N, K, lda, ldb, ldc, Hs, sAb, sAh, sBb, sBh, sCb, sCh, act);
  } else {
    if (edge) gemm_wmma_kernel<false, true><<<grid, 256, 0, st>>>(A, Bt, bias, C, M, N, K, lda, ldb, ldc, Hs, sAb, sAh, sBb, sBh, sCb, sCh, act);
    else      gemm_wmma_kernel<false, false><<<grid, 256, 0, st>>>(A, Bt, bias, C, M, N, K, lda, ldb, ldc, Hs, sAb, sAh, sBb, sBh, sCb, sCh, act);
  }
}

extern "C" void kernel_launch(void* const* d_in, const int* in_sizes, int n_in,
                              void* d_out, int out_size, void* d_ws, size_t ws_size,
                              hipStream_t stream) {
  (void)in_sizes; (void)n_in; (void)out_size; (void)ws_size;
  const int*   ids      = (const int*)d_in[0];
  const int*   amask    = (const int*)d_in[1];
  const float* extra    = (const float*)d_in[2];
  const float* word_emb = (const float*)d_in[3];
  const float* pos_emb  = (const float*)d_in[4];
  const float* type_emb = (const float*)d_in[5];
  const float* eg   = (const float*)d_in[6];
  const float* ebv  = (const float*)d_in[7];
  const float* Wq   = (const float*)d_in[8];
  const float* bq   = (const float*)d_in[9];
  const float* Wk   = (const float*)d_in[10];
  const float* bk   = (const float*)d_in[11];
  const float* Wv   = (const float*)d_in[12];
  const float* bv   = (const float*)d_in[13];
  const float* Wo   = (const float*)d_in[14];
  const float* bo   = (const float*)d_in[15];
  const float* ln1g = (const float*)d_in[16];
  const float* ln1b = (const float*)d_in[17];
  const float* W1   = (const float*)d_in[18];
  const float* b1   = (const float*)d_in[19];
  const float* W2   = (const float*)d_in[20];
  const float* b2   = (const float*)d_in[21];
  const float* ln2g = (const float*)d_in[22];
  const float* ln2b = (const float*)d_in[23];
  const float* c1w  = (const float*)d_in[24];
  const float* c1b  = (const float*)d_in[25];
  const float* c2w  = (const float*)d_in[26];
  const float* c2b  = (const float*)d_in[27];
  const float* c3w  = (const float*)d_in[28];
  const float* c3b  = (const float*)d_in[29];
  const float* We   = (const float*)d_in[30];
  const float* beP  = (const float*)d_in[31];
  const float* Wc   = (const float*)d_in[32];
  const float* bc   = (const float*)d_in[33];
  const float* WoutP= (const float*)d_in[34];
  const float* bout = (const float*)d_in[35];

  char* wp = (char*)d_ws;
  auto carve = [&](size_t bytes) -> void* {
    void* p = (void*)wp;
    wp += (bytes + 255) & ~(size_t)255;
    return p;
  };
  // slack rows so edge-tile over-reads stay in-buffer (results masked on store)
  const size_t QK_HALVES = (size_t)NTOK * DD + 32 * DD;
  const size_t P16_HALVES = (size_t)ATT_ROWS * SP + 32 * SP;
  const size_t VT_HALVES = (size_t)BB * HHN * DHD * SP;

  int*    pos   = (int*)carve((size_t)NTOK * 4);
  float*  x     = (float*)carve((size_t)NTOK * DD * 4);
  half_t* x16   = (half_t*)carve((size_t)NTOK * DD * 2);
  half_t* q16   = (half_t*)carve(QK_HALVES * 2);
  half_t* k16   = (half_t*)carve(QK_HALVES * 2);
  half_t* v16   = (half_t*)carve((size_t)NTOK * DD * 2);
  half_t* vT    = (half_t*)carve(VT_HALVES * 2);
  half_t* ctx16 = (half_t*)carve((size_t)NTOK * DD * 2);
  half_t* h16   = (half_t*)carve((size_t)NTOK * FFD * 2);
  float*  y32   = (float*)carve((size_t)NTOK * DD * 4);
  float*  att   = (float*)carve((size_t)ATT_ROWS * SP * 4);
  half_t* p16   = (half_t*)carve(P16_HALVES * 2);
  half_t* WqT   = (half_t*)carve((size_t)LLN * DD * DD * 2);
  half_t* WkT   = (half_t*)carve((size_t)LLN * DD * DD * 2);
  half_t* WvT   = (half_t*)carve((size_t)LLN * DD * DD * 2);
  half_t* WoT   = (half_t*)carve((size_t)LLN * DD * DD * 2);
  half_t* W1T   = (half_t*)carve((size_t)LLN * DD * FFD * 2);
  half_t* W2T   = (half_t*)carve((size_t)LLN * DD * FFD * 2);
  float*  c1    = (float*)carve((size_t)BB * 64 * 766 * 4);
  float*  p1    = (float*)carve((size_t)BB * 64 * 255 * 4);
  float*  c2    = (float*)carve((size_t)BB * 32 * 253 * 4);
  float*  p2    = (float*)carve((size_t)BB * 32 * 84 * 4);
  float*  c3    = (float*)carve((size_t)BB * 16 * 82 * 4);
  float*  p3    = (float*)carve((size_t)BB * 16 * 27 * 4);

  // Zero-fill K-padded f16 buffers (padding cols must be exact zeros)
  {
    long w1 = (long)(P16_HALVES * 2 / 4);
    zero_kernel<<<(w1 + 255) / 256, 256, 0, stream>>>((unsigned int*)p16, w1);
    long w2 = (long)(VT_HALVES * 2 / 4);
    zero_kernel<<<(w2 + 255) / 256, 256, 0, stream>>>((unsigned int*)vT, w2);
  }

  // Weight transpose + f16 conversion (deterministic, every call)
  {
    long t1 = (long)LLN * DD * DD;
    wt_f16_kernel<<<(t1 + 255) / 256, 256, 0, stream>>>(Wq, WqT, DD, DD, t1);
    wt_f16_kernel<<<(t1 + 255) / 256, 256, 0, stream>>>(Wk, WkT, DD, DD, t1);
    wt_f16_kernel<<<(t1 + 255) / 256, 256, 0, stream>>>(Wv, WvT, DD, DD, t1);
    wt_f16_kernel<<<(t1 + 255) / 256, 256, 0, stream>>>(Wo, WoT, DD, DD, t1);
    long t2 = (long)LLN * DD * FFD;
    wt_f16_kernel<<<(t2 + 255) / 256, 256, 0, stream>>>(W1, W1T, DD, FFD, t2);
    wt_f16_kernel<<<(t2 + 255) / 256, 256, 0, stream>>>(W2, W2T, FFD, DD, t2);
  }

  pos_kernel<<<(NTOK + 255) / 256, 256, 0, stream>>>(ids, pos);
  embed_ln_kernel<<<NTOK, 256, 0, stream>>>(ids, pos, word_emb, pos_emb, type_emb,
                                            eg, ebv, x, x16);

  const long sQKb = (long)SSQ * DD, sQKh = DHD;            // q16/k16 per (b,h)
  const long sAtth = (long)SSQ * SP, sAttb = (long)HHN * sAtth;
  const long sVTh = (long)DHD * SP, sVTb = (long)HHN * sVTh;

  for (int l = 0; l < LLN; l++) {
    const half_t* wqT = WqT + (size_t)l * DD * DD;
    const half_t* wkT = WkT + (size_t)l * DD * DD;
    const half_t* wvT = WvT + (size_t)l * DD * DD;
    const half_t* woT = WoT + (size_t)l * DD * DD;
    const half_t* w1T = W1T + (size_t)l * DD * FFD;
    const half_t* w2T = W2T + (size_t)l * DD * FFD;

    // Q, K, V projections (aligned, f16 out)
    launch_gemm(stream, true, false, x16, wqT, bq + l * DD, q16, NTOK, DD, DD,
                DD, DD, DD, 0, 1, 1, 0, 0, 0, 0, 0, 0);
    launch_gemm(stream, true, false, x16, wkT, bk + l * DD, k16, NTOK, DD, DD,
                DD, DD, DD, 0, 1, 1, 0, 0, 0, 0, 0, 0);
    launch_gemm(stream, true, false, x16, wvT, bv + l * DD, v16, NTOK, DD, DD,
                DD, DD, DD, 0, 1, 1, 0, 0, 0, 0, 0, 0);
    {
      long tv = (long)BB * HHN * DHD * SSQ;
      vT_kernel<<<(tv + 255) / 256, 256, 0, stream>>>(v16, vT);
    }
    // scores[b,h] = Q K^T  (edge stores; ldc = SP)
    launch_gemm(stream, false, true, q16, k16, nullptr, att, SSQ, SSQ, DHD,
                DD, DD, SP, 0, BB * HHN, HHN, sQKb, sQKh, sQKb, sQKh,
                sAttb, sAtth);
    softmax_mask_kernel<<<(int)ATT_ROWS, 256, 0, stream>>>(att, p16, amask, 0.125f);
    // ctx[b,:,h,:] = P V  (A = f16 probs, K padded to SP with zeros)
    launch_gemm(stream, true, true, p16, vT, nullptr, ctx16, SSQ, DHD, SP,
                SP, SP, DD, 0, BB * HHN, HHN, sAttb, sAtth, sVTb, sVTh,
                sQKb, sQKh);
    // attention output projection + residual LN
    launch_gemm(stream, false, false, ctx16, woT, bo + l * DD, y32, NTOK, DD, DD,
                DD, DD, DD, 0, 1, 1, 0, 0, 0, 0, 0, 0);
    add_ln_kernel<<<NTOK, 256, 0, stream>>>(x, y32, ln1g + l * DD, ln1b + l * DD,
                                            x, x16);
    // FFN (GELU fused into first GEMM, f16 intermediate)
    launch_gemm(stream, true, false, x16, w1T, b1 + l * FFD, h16, NTOK, FFD, DD,
                DD, DD, FFD, 1, 1, 1, 0, 0, 0, 0, 0, 0);
    launch_gemm(stream, false, false, h16, w2T, b2 + l * DD, y32, NTOK, DD, FFD,
                FFD, FFD, DD, 0, 1, 1, 0, 0, 0, 0, 0, 0);
    add_ln_kernel<<<NTOK, 256, 0, stream>>>(x, y32, ln2g + l * DD, ln2b + l * DD,
                                            x, x16);
  }

  // attn_last (strip SP padding) -> d_out[384:]
  copy_att_kernel<<<(long)(ATTN_ELEMS + 255) / 256, 256, 0, stream>>>(att,
      (float*)d_out + 384);

  // Conv head: x viewed as [B, C=200, W=768]
  {
    long t;
    t = (long)BB * 64 * 766;
    conv_relu_kernel<<<(t + 255) / 256, 256, 0, stream>>>(x, c1w, c1b, c1, BB, 200, 768, 64);
    t = (long)BB * 64 * 255;
    pool3_kernel<<<(t + 255) / 256, 256, 0, stream>>>(c1, p1, (long)BB * 64, 766);
    t = (long)BB * 32 * 253;
    conv_relu_kernel<<<(t + 255) / 256, 256, 0, stream>>>(p1, c2w, c2b, c2, BB, 64, 255, 32);
    t = (long)BB * 32 * 84;
    pool3_kernel<<<(t + 255) / 256, 256, 0, stream>>>(c2, p2, (long)BB * 32, 253);
    t = (long)BB * 16 * 82;
    conv_relu_kernel<<<(t + 255) / 256, 256, 0, stream>>>(p2, c3w, c3b, c3, BB, 32, 84, 16);
    t = (long)BB * 16 * 27;
    pool3_kernel<<<(t + 255) / 256, 256, 0, stream>>>(c3, p3, (long)BB * 16, 82);
  }
  head_kernel<<<BB, 128, 0, stream>>>(p3, extra, We, beP, Wc, bc, WoutP, bout,
                                      (float*)d_out);
}
